// AttnBlock_Fusion_21268678049861
// MI455X (gfx1250) — matrix operand order repaired and checked by reading, not compiled, not run on hardware
//
#include <hip/hip_runtime.h>

typedef __attribute__((ext_vector_type(16))) __bf16 v16bf;
typedef __attribute__((ext_vector_type(8)))  float  v8f;
typedef __attribute__((ext_vector_type(4)))  unsigned int u32x4;
typedef __attribute__((ext_vector_type(4)))  int i32x4;
typedef __attribute__((ext_vector_type(8)))  int i32x8;

#define DEVINL __device__ __forceinline__

#if defined(__has_builtin)
#  if __has_builtin(__builtin_amdgcn_tensor_load_to_lds)
#    define HAVE_TDM 1
#  endif
#endif
#ifndef HAVE_TDM
#  define HAVE_TDM 0
#endif

constexpr int NB   = 16;    // batch
constexpr int C    = 512;   // channels
constexpr int HW   = 1024;  // spatial
constexpr int NT   = NB * HW;      // 16384 (flattened b*hw)
constexpr int CPG  = 16;    // channels per group (512/32)

// GEMM tiling: block tile 128x64, K step 32, 256 threads = 8 waves,
// wave grid 4(M) x 2(N), each wave -> 32x32 = 2x2 wmma 16x16 tiles.
// Double-buffered LDS so the TDM overlaps with WMMA.
constexpr int BM = 128, BN = 64, BK = 32, LPAD = 8, LK = BK + LPAD; // 40

DEVINL unsigned short f32_to_bf16(float f) {
  unsigned int u = __float_as_uint(f);
  u += 0x7FFFu + ((u >> 16) & 1u);          // round-to-nearest-even
  return (unsigned short)(u >> 16);
}

DEVINL unsigned lds_addr_of(const void* p) {
  return (unsigned)(unsigned long long)(uintptr_t)p;  // generic LDS ptr low 32 = LDS offset
}

union Frag { unsigned int u[8]; v16bf v; };

#if HAVE_TDM
// ---------------------------------------------------------------------------
// Tensor Data Mover: 2D tile load global->LDS with hardware row padding.
// D# per CDNA5 ISA:
//   group0: count=1 | lds_addr | global_addr | type=2
//   group1: data_size=2B, pad_enable, pad_interval=3 (16 DWORD = 64B row),
//           pad_amount=3 (4 DWORD = 8 bf16) -> LDS rows of 40 ushorts,
//           tensor_dim0/1, tile_dim0/1, tensor_dim0_stride
// Issued by one wave; completion via TENSORcnt.
// ---------------------------------------------------------------------------
DEVINL void tdm_load_2d(unsigned lds_addr, const void* gptr,
                        unsigned tile_x, unsigned tile_y,
                        unsigned long long row_stride_elems) {
  unsigned long long ga = (unsigned long long)(uintptr_t)gptr;
  unsigned long long tw = row_stride_elems;        // tensor_dim0 (OOB bound)
  unsigned long long th = tile_y;                  // tensor_dim1
  u32x4 g0;
  g0[0] = 1u;                                      // count=1, user descriptor
  g0[1] = lds_addr;                                // LDS byte address
  g0[2] = (unsigned)(ga & 0xFFFFFFFFu);            // global_addr[31:0]
  g0[3] = (unsigned)((ga >> 32) & 0x01FFFFFFu) | (2u << 30);  // [56:32] | type=2
  unsigned w0 = (1u << 16)                         // data_size = 2 bytes
              | (1u << 20)                         // pad_enable
              | (3u << 22)                         // pad_interval: 16 DWORDs
              | (3u << 25);                        // pad_amount:   4 DWORDs
  i32x8 g1;
  g1[0] = (int)w0;
  g1[1] = (int)((unsigned)(tw & 0xFFFFu) << 16);   // [15:0] atomic addr=0, [31:16] dim0 lo
  g1[2] = (int)(((tw >> 16) & 0xFFFFu) | ((th & 0xFFFFu) << 16));
  g1[3] = (int)(((th >> 16) & 0xFFFFu) | (tile_x << 16));   // tile_dim0
  g1[4] = (int)(tile_y & 0xFFFFu);                 // tile_dim1, tile_dim2=0
  g1[5] = (int)(row_stride_elems & 0xFFFFFFFFu);   // tensor_dim0_stride lo
  g1[6] = (int)((row_stride_elems >> 32) & 0xFFFFu);        // stride hi | dim1_stride=0
  g1[7] = 0;
  i32x4 z4 = {0, 0, 0, 0};
#if __clang_major__ >= 23
  i32x8 z8 = {0, 0, 0, 0, 0, 0, 0, 0};
  __builtin_amdgcn_tensor_load_to_lds(g0, g1, z4, z4, z8, 0);
#else
  __builtin_amdgcn_tensor_load_to_lds(g0, g1, z4, z4, 0);
#endif
}
#endif // HAVE_TDM

// ---------------------------------------------------------------------------
// Weight conversion f32 -> bf16 (4 matrices of 512x512, row-major [o][c])
// ---------------------------------------------------------------------------
__global__ __launch_bounds__(256)
void k_convert_w(const float* __restrict__ w0, const float* __restrict__ w1,
                 const float* __restrict__ w2, const float* __restrict__ w3,
                 unsigned short* __restrict__ o0, unsigned short* __restrict__ o1,
                 unsigned short* __restrict__ o2, unsigned short* __restrict__ o3) {
  int gid = blockIdx.x * 256 + threadIdx.x;     // 4 * 2^18 total
  int mat = gid >> 18;
  int idx = gid & 0x3FFFF;
  const float* w = (mat == 0) ? w0 : (mat == 1) ? w1 : (mat == 2) ? w2 : w3;
  unsigned short* o = (mat == 0) ? o0 : (mat == 1) ? o1 : (mat == 2) ? o2 : o3;
  o[idx] = f32_to_bf16(w[idx]);
}

// ---------------------------------------------------------------------------
// GroupNorm (32 groups, eps 1e-6) -> bf16, output layout [c][b*HW + p]
// ---------------------------------------------------------------------------
__global__ __launch_bounds__(256)
void k_groupnorm(const float* __restrict__ x, const float* __restrict__ gamma,
                 const float* __restrict__ beta, unsigned short* __restrict__ out) {
  int bg = blockIdx.x;
  int b = bg >> 5, g = bg & 31;
  int c0 = g * CPG;
  const float* xp = x + ((size_t)b * C + c0) * HW;
  int tid = threadIdx.x;

  float s = 0.f, ss = 0.f;
  for (int c = 0; c < CPG; ++c) {
    const float* row = xp + (size_t)c * HW;
    #pragma unroll
    for (int it = 0; it < HW / 256; ++it) {
      float v = row[tid + 256 * it];
      s += v; ss += v * v;
    }
  }
  __shared__ float rs[8], rss[8], stat[2];
  #pragma unroll
  for (int m = 16; m >= 1; m >>= 1) {
    s  += __shfl_xor(s,  m, 32);
    ss += __shfl_xor(ss, m, 32);
  }
  if ((tid & 31) == 0) { rs[tid >> 5] = s; rss[tid >> 5] = ss; }
  __syncthreads();
  if (tid == 0) {
    float S = 0.f, SS = 0.f;
    #pragma unroll
    for (int i = 0; i < 8; ++i) { S += rs[i]; SS += rss[i]; }
    float inv_n = 1.0f / (float)(CPG * HW);
    float mean = S * inv_n;
    float var  = SS * inv_n - mean * mean;
    stat[0] = mean;
    stat[1] = rsqrtf(var + 1e-6f);
  }
  __syncthreads();
  float mean = stat[0], rstd = stat[1];

  for (int c = 0; c < CPG; ++c) {
    float gm = gamma[c0 + c], bt = beta[c0 + c];
    const float* row = xp + (size_t)c * HW;
    unsigned short* orow = out + (size_t)(c0 + c) * NT + (size_t)b * HW;
    #pragma unroll
    for (int it = 0; it < HW / 256; ++it) {
      int p = tid + 256 * it;
      orow[p] = f32_to_bf16((row[p] - mean) * rstd * gm + bt);
    }
  }
}

// ---------------------------------------------------------------------------
// Generic bf16 WMMA GEMM (double-buffered):  D = A * B  (+epilogue)
//   A: bf16 row-major [M][K], stride lda, batch stride a_bs (elements)
//   B: if !B_PRE_T: bf16 row-major [K][N], stride ldb  (transposed into LDS)
//      if  B_PRE_T: bf16 row-major [N][K], stride ldb  (TDM straight to LDS)
// OUT_MODE 0: bf16 Out[m*out_ld+n] = D + bias[m]        (bias may be null)
// OUT_MODE 1: bf16 Out[n*out_ld+m] = D + bias[m]        (transposed store)
// OUT_MODE 2: f32  Out[m*out_ld+n] = D * scale
// OUT_MODE 3: f32  d_out[((n/HW)*C+m)*HW + n%HW] = resid[idx] + bias[m] + D
// ---------------------------------------------------------------------------
template <int OUT_MODE, bool B_PRE_T>
__global__ __launch_bounds__(256)
void k_gemm_wmma(const unsigned short* __restrict__ A, int lda, size_t a_bs,
                 const unsigned short* __restrict__ B, int ldb, size_t b_bs,
                 void* __restrict__ Out, int out_ld, size_t out_bs,
                 const float* __restrict__ bias, const float* __restrict__ resid,
                 float scale, int K) {
  __shared__ unsigned short As[2][BM][LK];
  __shared__ unsigned short Bs[2][BN][LK];

  int bz = blockIdx.z;
  const unsigned short* Ab = A + a_bs * bz;
  const unsigned short* Bb = B + b_bs * bz;
  int m0 = blockIdx.y * BM;
  int n0 = blockIdx.x * BN;

  int tid  = threadIdx.x;
  int lane = tid & 31;
  int wave = tid >> 5;
  int wm = (wave >> 1) * 32;   // 0,32,64,96
  int wn = (wave & 1) * 32;    // 0,32

  const int niter = K / BK;
  constexpr int TDM_OPS = B_PRE_T ? 2 : 1;   // TDM issues per stage (wave 0)

  // staging-thread decode
  int pos = tid * 8;
  int ar = pos >> 5, ac = pos & 31;          // A: 64 rows/pass x 32 cols
  int bkk = tid >> 3;                        // B !pre-T: k row 0..31
  int bnn = (tid & 7) * 8;                   //           n col 0..56

  v8f acc[2][2];
  #pragma unroll
  for (int i = 0; i < 2; ++i)
    #pragma unroll
    for (int j = 0; j < 2; ++j) {
      v8f z = {0.f, 0.f, 0.f, 0.f, 0.f, 0.f, 0.f, 0.f};
      acc[i][j] = z;
    }

  // fragment lane decode (ISA 7.12.2, 16-bit A 16x32 / B 32x16 / f32 C 16x16)
  int am  = lane & 15;
  int akb = (lane < 16) ? 0 : 8;
  int bn  = lane & 15;
  int bkb = (lane < 16) ? 0 : 8;

  // ---- prologue: stage tile 0 into buffer 0 ----
#if HAVE_TDM
  if (wave == 0) { // uniform branch: one wave drives the TDM
    tdm_load_2d(lds_addr_of(&As[0][0][0]), Ab + (size_t)m0 * lda, BK, BM,
                (unsigned long long)lda);
    if (B_PRE_T)
      tdm_load_2d(lds_addr_of(&Bs[0][0][0]), Bb + (size_t)n0 * ldb, BK, BN,
                  (unsigned long long)ldb);
  }
#else
  *(uint4*)&As[0][ar][ac]      = *(const uint4*)(Ab + (size_t)(m0 + ar) * lda + ac);
  *(uint4*)&As[0][ar + 64][ac] = *(const uint4*)(Ab + (size_t)(m0 + ar + 64) * lda + ac);
  if (B_PRE_T)
    *(uint4*)&Bs[0][ar][ac] = *(const uint4*)(Bb + (size_t)(n0 + ar) * ldb + ac);
#endif
  if (!B_PRE_T) {
    uint4 v = *(const uint4*)(Bb + (size_t)bkk * ldb + n0 + bnn);
    unsigned short tmp[8];
    *(uint4*)tmp = v;
    #pragma unroll
    for (int i = 0; i < 8; ++i) Bs[0][bnn + i][bkk] = tmp[i];
  }

  for (int it = 0; it < niter; ++it) {
    int cb = it & 1, nbuf = cb ^ 1;
    int k0 = it * BK;
    bool has_next = (it + 1) < niter;
    int kn = k0 + BK;

    // ---- issue stage(i+1); only wait for stage(i) ----
#if HAVE_TDM
    if (wave == 0) {
      if (has_next) {
        tdm_load_2d(lds_addr_of(&As[nbuf][0][0]), Ab + (size_t)m0 * lda + kn,
                    BK, BM, (unsigned long long)lda);
        if (B_PRE_T)
          tdm_load_2d(lds_addr_of(&Bs[nbuf][0][0]), Bb + (size_t)n0 * ldb + kn,
                      BK, BN, (unsigned long long)ldb);
        __builtin_amdgcn_s_wait_tensorcnt(TDM_OPS); // tile i done, i+1 in flight
      } else {
        __builtin_amdgcn_s_wait_tensorcnt(0);
      }
    }
    uint4 bR;
    if (!B_PRE_T && has_next) {
      const unsigned short* s = Bb + (size_t)(kn + bkk) * ldb + n0 + bnn;
      bR = *(const uint4*)s;
      if (it + 2 < niter) __builtin_prefetch(s + BK, 0, 0);  // global_prefetch_b8
    }
#else
    uint4 aR0, aR1, bR, bR2;
    if (has_next) {
      aR0 = *(const uint4*)(Ab + (size_t)(m0 + ar) * lda + kn + ac);
      aR1 = *(const uint4*)(Ab + (size_t)(m0 + ar + 64) * lda + kn + ac);
      if (B_PRE_T) bR2 = *(const uint4*)(Bb + (size_t)(n0 + ar) * ldb + kn + ac);
    }
    if (!B_PRE_T && has_next) {
      const unsigned short* s = Bb + (size_t)(kn + bkk) * ldb + n0 + bnn;
      bR = *(const uint4*)s;
      if (it + 2 < niter) __builtin_prefetch(s + BK, 0, 0);
    }
#endif
    __syncthreads();   // buffer cb fully resident for all waves

    // ---- compute on buffer cb ----
    Frag a[2], bf[2];
    #pragma unroll
    for (int t = 0; t < 2; ++t) {
      int mrow = wm + t * 16 + am;
      #pragma unroll
      for (int j = 0; j < 8; ++j) {
        int kk = akb + ((j >> 2) << 4) + ((j & 3) << 1);
        a[t].u[j] = *(const unsigned int*)&As[cb][mrow][kk];
      }
      int nrow = wn + t * 16 + bn;
      #pragma unroll
      for (int j = 0; j < 8; ++j) {
        int kk = bkb + ((j >> 2) << 4) + ((j & 3) << 1);
        bf[t].u[j] = *(const unsigned int*)&Bs[cb][nrow][kk];
      }
    }
    #pragma unroll
    for (int tm = 0; tm < 2; ++tm)
      #pragma unroll
      for (int tn = 0; tn < 2; ++tn)
        acc[tm][tn] = __builtin_amdgcn_wmma_f32_16x16x32_bf16(
            false, a[tm].v, false, bf[tn].v, (short)0, acc[tm][tn], false, false);

    // ---- commit staged tile(i+1) into buffer nbuf ----
#if !HAVE_TDM
    if (has_next) {
      *(uint4*)&As[nbuf][ar][ac]      = aR0;
      *(uint4*)&As[nbuf][ar + 64][ac] = aR1;
      if (B_PRE_T) *(uint4*)&Bs[nbuf][ar][ac] = bR2;
    }
#endif
    if (!B_PRE_T && has_next) {
      unsigned short tmp[8];
      *(uint4*)tmp = bR;
      #pragma unroll
      for (int i = 0; i < 8; ++i) Bs[nbuf][bnn + i][bkk] = tmp[i];
    }
    __syncthreads();   // everyone done reading cb before it is overwritten
  }

  // epilogue
  int cn  = lane & 15;
  int cmb = (lane < 16) ? 0 : 8;
  #pragma unroll
  for (int tm = 0; tm < 2; ++tm) {
    #pragma unroll
    for (int tn = 0; tn < 2; ++tn) {
      #pragma unroll
      for (int r = 0; r < 8; ++r) {
        int m = m0 + wm + tm * 16 + cmb + r;
        int n = n0 + wn + tn * 16 + cn;
        float val = acc[tm][tn][r];
        if (OUT_MODE == 0) {
          float bv = bias ? bias[m] : 0.f;
          ((unsigned short*)Out)[out_bs * bz + (size_t)m * out_ld + n] =
              f32_to_bf16(val + bv);
        } else if (OUT_MODE == 1) {
          ((unsigned short*)Out)[out_bs * bz + (size_t)n * out_ld + m] =
              f32_to_bf16(val + bias[m]);
        } else if (OUT_MODE == 2) {
          ((float*)Out)[out_bs * bz + (size_t)m * out_ld + n] = val * scale;
        } else {
          int bb = n >> 10, p = n & 1023;
          size_t idx = ((size_t)bb * C + m) * HW + p;
          ((float*)Out)[idx] = resid[idx] + bias[m] + val;
        }
      }
    }
  }
}

// ---------------------------------------------------------------------------
// Row softmax: one block per row (16384 rows of 1024 f32), write bf16 P[i][j]
// ---------------------------------------------------------------------------
__global__ __launch_bounds__(256)
void k_softmax(const float* __restrict__ S, unsigned short* __restrict__ P) {
  int row = blockIdx.x;
  int tid = threadIdx.x;
  const float* sr = S + (size_t)row * HW;

  float v[4];
  float mx = -3.0e38f;
  #pragma unroll
  for (int i = 0; i < 4; ++i) { v[i] = sr[tid + 256 * i]; mx = fmaxf(mx, v[i]); }

  __shared__ float redmax[8], redsum[8], bcast[2];
  #pragma unroll
  for (int m = 16; m >= 1; m >>= 1) mx = fmaxf(mx, __shfl_xor(mx, m, 32));
  if ((tid & 31) == 0) redmax[tid >> 5] = mx;
  __syncthreads();
  if (tid == 0) {
    float t = redmax[0];
    #pragma unroll
    for (int i = 1; i < 8; ++i) t = fmaxf(t, redmax[i]);
    bcast[0] = t;
  }
  __syncthreads();
  float bm = bcast[0];

  float s = 0.f;
  #pragma unroll
  for (int i = 0; i < 4; ++i) { v[i] = __expf(v[i] - bm); s += v[i]; }
  #pragma unroll
  for (int m = 16; m >= 1; m >>= 1) s += __shfl_xor(s, m, 32);
  if ((tid & 31) == 0) redsum[tid >> 5] = s;
  __syncthreads();
  if (tid == 0) {
    float t = 0.f;
    #pragma unroll
    for (int i = 0; i < 8; ++i) t += redsum[i];
    bcast[1] = 1.0f / t;
  }
  __syncthreads();
  float inv = bcast[1];

  unsigned short* pr = P + (size_t)row * HW;
  #pragma unroll
  for (int i = 0; i < 4; ++i) pr[tid + 256 * i] = f32_to_bf16(v[i] * inv);
}

// ---------------------------------------------------------------------------
// launch
// ---------------------------------------------------------------------------
extern "C" void kernel_launch(void* const* d_in, const int* in_sizes, int n_in,
                              void* d_out, int out_size, void* d_ws, size_t ws_size,
                              hipStream_t stream) {
  (void)in_sizes; (void)n_in; (void)out_size; (void)ws_size;
  const float* x    = (const float*)d_in[0];
  const float* y    = (const float*)d_in[1];
  const float* nsc  = (const float*)d_in[2];
  const float* nbi  = (const float*)d_in[3];
  const float* n1sc = (const float*)d_in[4];
  const float* n1bi = (const float*)d_in[5];
  const float* wq   = (const float*)d_in[6];
  const float* bq   = (const float*)d_in[7];
  const float* wk   = (const float*)d_in[8];
  const float* bk   = (const float*)d_in[9];
  const float* wv   = (const float*)d_in[10];
  const float* bv   = (const float*)d_in[11];
  const float* wp   = (const float*)d_in[12];
  const float* bp   = (const float*)d_in[13];
  float* out = (float*)d_out;

  // workspace carve-up (256B aligned)
  char* ws = (char*)d_ws;
  size_t off = 0;
  auto carve = [&](size_t bytes) {
    char* p = ws + off;
    off += (bytes + 255) & ~(size_t)255;
    return p;
  };
  unsigned short* hx  = (unsigned short*)carve((size_t)C * NT * 2);   // GN(x) [c][n]
  unsigned short* hy  = (unsigned short*)carve((size_t)C * NT * 2);   // GN(y) [c][n]
  unsigned short* wqb = (unsigned short*)carve((size_t)C * C * 2);
  unsigned short* wkb = (unsigned short*)carve((size_t)C * C * 2);
  unsigned short* wvb = (unsigned short*)carve((size_t)C * C * 2);
  unsigned short* wpb = (unsigned short*)carve((size_t)C * C * 2);
  unsigned short* qT  = (unsigned short*)carve((size_t)NT * C * 2);   // q  [n][c]
  unsigned short* kb_ = (unsigned short*)carve((size_t)C * NT * 2);   // k  [c][n]
  unsigned short* vb_ = (unsigned short*)carve((size_t)C * NT * 2);   // v  [c][n]
  float*          Sb  = (float*)         carve((size_t)NB * HW * HW * 4); // scores
  unsigned short* Pb  = (unsigned short*)carve((size_t)NB * HW * HW * 2); // softmax
  unsigned short* ao  = (unsigned short*)carve((size_t)C * NT * 2);   // attn out [c][n]

  dim3 blk(256);

  // 1) weights -> bf16
  k_convert_w<<<dim3(4 * C * C / 256), blk, 0, stream>>>(wq, wk, wv, wp,
                                                         wqb, wkb, wvb, wpb);
  // 2) group norms
  k_groupnorm<<<dim3(NB * 32), blk, 0, stream>>>(x, nsc, nbi, hx);
  k_groupnorm<<<dim3(NB * 32), blk, 0, stream>>>(y, n1sc, n1bi, hy);

  // 3) k = wk*hx, v = wv*hx (bf16 [c][n]); q = wq*hy transposed -> qT [n][c]
  dim3 g_qkv(NT / BN, C / BM, 1);
  k_gemm_wmma<0, false><<<g_qkv, blk, 0, stream>>>(wkb, C, 0, hx, NT, 0,
      kb_, NT, 0, bk, nullptr, 1.f, C);
  k_gemm_wmma<0, false><<<g_qkv, blk, 0, stream>>>(wvb, C, 0, hx, NT, 0,
      vb_, NT, 0, bv, nullptr, 1.f, C);
  k_gemm_wmma<1, false><<<g_qkv, blk, 0, stream>>>(wqb, C, 0, hy, NT, 0,
      qT, C, 0, bq, nullptr, 1.f, C);

  // 4) scores S[b][i][j] = (qT[b] * k[b]) / sqrt(C), batched over z
  const float sc = 0.044194173824159216f; // 512^-0.5
  dim3 g_sc(HW / BN, HW / BM, NB);
  k_gemm_wmma<2, false><<<g_sc, blk, 0, stream>>>(
      qT, C, (size_t)HW * C,            // A = qT rows i, batch stride HW*C
      kb_, NT, (size_t)HW,              // B = k[c][n], batch offset = columns
      Sb, HW, (size_t)HW * HW, nullptr, nullptr, sc, C);

  // 5) softmax rows -> bf16 P[i][j]
  k_softmax<<<dim3(NB * HW), blk, 0, stream>>>(Sb, Pb);

  // 6) attn_out[c][i] = sum_j v[c][j] * P[i][j]  (P is pre-transposed B -> TDM)
  dim3 g_av(HW / BN, C / BM, NB);
  k_gemm_wmma<0, true><<<g_av, blk, 0, stream>>>(
      vb_, NT, (size_t)HW,              // A = v[c][j]
      Pb, HW, (size_t)HW * HW,          // B = P [n=i][k=j]
      ao, NT, (size_t)HW, nullptr, nullptr, 1.f, HW);

  // 7) d_out = x + wp*attn_out + bp   (f32)
  dim3 g_fp(NT / BN, C / BM, 1);
  k_gemm_wmma<3, false><<<g_fp, blk, 0, stream>>>(
      wpb, C, 0, ao, NT, 0, out, NT, 0, bp, x, 1.f, C);
}
